// SharedEncoder_67267777789974
// MI455X (gfx1250) — matrix operand order repaired and checked by reading, not compiled, not run on hardware
//
#include <hip/hip_runtime.h>
#include <math.h>

// ---------------- problem constants ----------------
#define BB   512
#define CC   102      // sensor channels
#define LL   168
#define FF   32       // filters
#define TT   7
#define CPAD 112      // 7*16
#define FEAT 64       // 2F
#define KFC  224      // F*T

// ---------------- WMMA types / helpers ----------------
typedef __attribute__((ext_vector_type(16))) __bf16 v16bf;
typedef __attribute__((ext_vector_type(8)))  float  v8f;

union Frag { v16bf v; uint4 q[2]; unsigned short h[16]; };

__device__ __forceinline__ v8f wmma_bf16(v16bf a, v16bf b, v8f c) {
  return __builtin_amdgcn_wmma_f32_16x16x32_bf16(false, a, false, b, (short)0, c,
                                                 false, false);
}

__device__ __forceinline__ unsigned short f2bf(float f) {
  unsigned u = __builtin_bit_cast(unsigned, f);
  unsigned r = u + 0x7fffu + ((u >> 16) & 1u);   // RNE
  return (unsigned short)(r >> 16);
}
__device__ __forceinline__ float bf2f(unsigned short h) {
  unsigned u = ((unsigned)h) << 16;
  return __builtin_bit_cast(float, u);
}

// ---------------- workspace layout (bytes) ----------------
#define OFF_FRAGS 0ll
#define OFF_XT    65536ll
#define SZ_XT     ((long long)BB*CC*LL*4)                 // 35,094,528
#define OFF_H2    (OFF_XT + SZ_XT)
#define SZ_H2     ((long long)BB*CC*39*FF*2)              // 130,351,104
#define OFF_H3    (OFF_H2 + SZ_H2)
#define SZ_H3     ((long long)BB*CC*18*FF*2)              // 60,162,048
#define OFF_H4    (OFF_H3 + SZ_H3)
#define SZ_H4     ((long long)BB*CC*TT*FF*2)              // 23,396,352
#define OFF_R     (OFF_H4 + SZ_H4)
#define SZ_R      ((long long)BB*CC*KFC*2)                // 23,396,352

// =====================================================================
// 0) transpose x (B,1,L,C) f32 -> xT (B*C, L) f32   (coalesced reads on C)
// =====================================================================
__global__ void __launch_bounds__(256) k_transpose(const float* __restrict__ x,
                                                   float* __restrict__ xT) {
  int idx = blockIdx.x * 256 + threadIdx.x;            // (b*L + l)*C + c
  if (idx >= BB * LL * CC) return;
  int c = idx % CC;
  int l = (idx / CC) % LL;
  int b = idx / (CC * LL);
  xT[((long long)(b * CC + c)) * LL + l] = x[idx];
}

// =====================================================================
// 1) repack weights into WMMA-fragment-ready bf16 buffers
// =====================================================================
__global__ void __launch_bounds__(256) k_repack(
    const float* __restrict__ w2, const float* __restrict__ w3,
    const float* __restrict__ w4, const float* __restrict__ wq,
    const float* __restrict__ wk, const float* __restrict__ wv,
    const float* __restrict__ wfc, unsigned short* __restrict__ frags) {
  unsigned short* fconv = frags;                 // [3][5][2][32][16]
  unsigned short* fqkv  = frags + 15360;         // [3][2][32][16]
  unsigned short* ffc   = frags + 18432;         // [7][4][32][16]
  const float* wl[3] = {w2, w3, w4};
  // conv B-fragments: B[K=fin][N=fout] = w[fout][fin][s]
  for (int i = threadIdx.x; i < 3 * 5 * 2 * 32 * 16; i += 256) {
    int e = i & 15, lane = (i >> 4) & 31, nt = (i >> 9) & 1;
    int s = (i >> 10) % 5, layer = (i >> 10) / 5;
    int half = lane >> 4, fo = (lane & 15) + 16 * nt, fi = (e & 7) + 8 * half + 16 * (e >> 3);
    fconv[i] = f2bf(wl[layer][(fo * FF + fi) * 5 + s]);
  }
  // qkv A-fragments: A[M=row][K=col] = W[row][col]
  const float* wm[3] = {wq, wk, wv};
  for (int i = threadIdx.x; i < 3 * 2 * 32 * 16; i += 256) {
    int e = i & 15, lane = (i >> 4) & 31, mt = (i >> 9) & 1, g = i >> 10;
    int half = lane >> 4;
    int row = (lane & 15) + 16 * mt;
    int col = e + 16 * half;
    fqkv[i] = f2bf(wm[g][row * FF + col]);
  }
  // fc B-fragments: B[K=m][N=o] = wfc[o][m]
  for (int i = threadIdx.x; i < 7 * 4 * 32 * 16; i += 256) {
    int e = i & 15, lane = (i >> 4) & 31, nt = (i >> 9) & 3, ks = i >> 11;
    int half = lane >> 4;
    int o  = (lane & 15) + 16 * nt;
    int mk = (e & 7) + 8 * half + 16 * (e >> 3) + 32 * ks;
    ffc[i] = f2bf(wfc[o * KFC + mk]);
  }
}

// =====================================================================
// shared conv GEMM: per-wave (b,c) slice in LDS  ->  relu(conv)  bf16 out
// =====================================================================
__device__ __forceinline__ void conv_gemm(const unsigned short* hrows, int LOUT,
                                          int MT, const unsigned short* fw,
                                          const float* bias,
                                          unsigned short* outp, int lane) {
  const int half = lane >> 4, lm = lane & 15;
  for (int nt = 0; nt < 2; ++nt) {
    Frag bw[5];
    for (int s = 0; s < 5; ++s) {
      const uint4* p = (const uint4*)(fw + ((s * 2 + nt) * 32 + lane) * 16);
      bw[s].q[0] = p[0];
      bw[s].q[1] = p[1];
    }
    const int fo = lm + 16 * nt;
    const float bb = bias[fo];
    for (int mt = 0; mt < MT; ++mt) {
      v8f acc = {0.f, 0.f, 0.f, 0.f, 0.f, 0.f, 0.f, 0.f};
      for (int s = 0; s < 5; ++s) {
        Frag a;
        const uint4* row = (const uint4*)(hrows + (2 * (16 * mt + lm) + s) * 32);
        a.q[0] = row[half];
        a.q[1] = row[2 + half];
        acc = wmma_bf16(a.v, bw[s].v, acc);
      }
      for (int r = 0; r < 8; ++r) {
        int t = r + 8 * half + 16 * mt;
        if (t < LOUT) {
          float v = fmaxf(acc[r] + bb, 0.f);
          outp[t * FF + fo] = f2bf(v);
        }
      }
    }
  }
}

// =====================================================================
// 2) fused conv1 (scalar, Cin=1) + conv2 (WMMA).  1 wave = 1 (b,c) slice.
// =====================================================================
__global__ void __launch_bounds__(128) k_conv12(
    const float* __restrict__ xT, const float* __restrict__ w1,
    const float* __restrict__ b1, const float* __restrict__ b2,
    const unsigned short* __restrict__ frags, unsigned short* __restrict__ h2) {
  __shared__ float xs[4][LL];
  __shared__ __align__(16) unsigned short h1s[4][100][FF];   // rows>=82 zero pad
  const int wave = threadIdx.x >> 5, lane = threadIdx.x & 31;
  const long long pair = (long long)blockIdx.x * 4 + wave;   // b*C + c
  const float* xrow = xT + pair * LL;
  for (int l = lane; l < LL; l += 32) xs[wave][l] = xrow[l];
  float w1r[5];
  for (int k = 0; k < 5; ++k) w1r[k] = w1[lane * 5 + k];
  const float b1r = b1[lane];
  __syncthreads();
  for (int t = 0; t < 100; ++t) {            // h1: 82 valid rows + zero pad
    float a = 0.f;
    if (t < 82) {
      a = b1r;
      for (int k = 0; k < 5; ++k) a += w1r[k] * xs[wave][2 * t + k];
      a = fmaxf(a, 0.f);
    }
    h1s[wave][t][lane] = f2bf(a);
  }
  __syncthreads();
  conv_gemm(&h1s[wave][0][0], 39, 3, frags, b2, h2 + pair * 39 * FF, lane);
}

// =====================================================================
// 3) generic bf16 conv layer (conv3 / conv4).  1 wave = 1 (b,c) slice.
//    Input slice staged with ASYNC memory->LDS DMA (ASYNCcnt-tracked),
//    no VGPR round-trip; zero-padding stores overlap the DMA.
// =====================================================================
template <int LIN, int LOUT, int MT, int LAYER>
__global__ void __launch_bounds__(128) k_conv(
    const unsigned short* __restrict__ in, unsigned short* __restrict__ out,
    const unsigned short* __restrict__ frags, const float* __restrict__ bias) {
  constexpr int ROWS = 32 * MT + 4;
  __shared__ __align__(16) unsigned short hs[4][ROWS][FF];
  const int wave = threadIdx.x >> 5, lane = threadIdx.x & 31;
  const long long pair = (long long)blockIdx.x * 4 + wave;

  // async DMA of the (b,c) slice straight into this wave's LDS region
  // (LDS aperture address = low 32 bits of the generic pointer, ISA 10.2)
  unsigned ldsbase = (unsigned)(unsigned long long)(uintptr_t)&hs[wave][0][0];
  unsigned gbase = (unsigned)(pair * (LIN * FF) * 2);     // byte offset, <2^31
  for (int i4 = lane; i4 < (LIN * 16) / 4; i4 += 32) {    // LIN*16 % 4 == 0
    unsigned ldsa = ldsbase + i4 * 16u;
    unsigned voff = gbase + i4 * 16u;
    asm volatile("global_load_async_to_lds_b128 %0, %1, %2"
                 :: "v"(ldsa), "v"(voff), "s"(in) : "memory");
  }
  // zero-pad rows [LIN, ROWS)
  unsigned* dst = (unsigned*)&hs[wave][0][0];
  for (int i = lane; i < (ROWS - LIN) * 16; i += 32) dst[LIN * 16 + i] = 0u;
  asm volatile("s_wait_asynccnt 0" ::: "memory");
  __syncthreads();

  conv_gemm(&hs[wave][0][0], LOUT, MT, frags + LAYER * 5120, bias,
            out + pair * LOUT * FF, lane);
}

// =====================================================================
// 4) attention: one workgroup (8 waves) per (b,t)
// =====================================================================
__global__ void __launch_bounds__(256) k_attn(
    const unsigned short* __restrict__ h4, const unsigned short* __restrict__ fqkv,
    const float* __restrict__ bq, const float* __restrict__ bk,
    const float* __restrict__ bv, const float* __restrict__ gamma_p,
    unsigned short* __restrict__ R) {
  __shared__ __align__(16) unsigned short xts[CPAD][FF];   // xt[c][f]
  __shared__ __align__(16) unsigned short qs[CPAD][FF];    // q[c][fq]
  __shared__ __align__(16) unsigned short kk[CPAD][FF];    // k[c][fq]
  __shared__ __align__(16) unsigned short vs[FF][128];     // v[f][i]
  __shared__ __align__(16) unsigned short sb[CPAD][128];   // scores^T / beta[j][i]
  const int tid = threadIdx.x, wave = tid >> 5, lane = tid & 31;
  const int half = lane >> 4, lm = lane & 15;
  const int b = blockIdx.x / TT, t = blockIdx.x % TT;

  // --- load xt (pad channels >= 102 with zeros) ---
  unsigned* xd = (unsigned*)&xts[0][0];
  for (int i = tid; i < CPAD * 16; i += 256) {
    int c = i >> 4, j = i & 15;
    unsigned v = 0u;
    if (c < CC)
      v = ((const unsigned*)h4)[((long long)(b * CC + c) * TT + t) * 16 + j];
    xd[i] = v;
  }
  __syncthreads();

  // --- q,k,v = W @ xt + bias : 3 * 2mt * 7nt = 42 WMMA tiles ---
  const float* biases[3] = {bq, bk, bv};
  for (int job = wave; job < 42; job += 8) {
    int g = job / 14, rem = job % 14, mt = rem / 7, nt = rem % 7;
    Frag a, bf;
    const uint4* pa = (const uint4*)(fqkv + ((g * 2 + mt) * 32 + lane) * 16);
    a.q[0] = pa[0];
    a.q[1] = pa[1];
    const int cc = lm + 16 * nt;
    const uint4* row = (const uint4*)&xts[cc][0];
    bf.q[0] = row[2 * half];
    bf.q[1] = row[2 * half + 1];
    v8f acc = {0.f, 0.f, 0.f, 0.f, 0.f, 0.f, 0.f, 0.f};
    acc = wmma_bf16(a.v, bf.v, acc);
    const float* bias = biases[g];
    for (int r = 0; r < 8; ++r) {
      int m = r + 8 * half + 16 * mt;
      unsigned short hv = f2bf(acc[r] + bias[m]);
      if (g == 0)      qs[cc][m] = hv;
      else if (g == 1) kk[cc][m] = hv;
      else             vs[m][cc] = hv;
    }
  }
  __syncthreads();

  // --- scores[i][j] = sum_f q[f][i] k[f][j]  (store transposed sb[j][i]) ---
  for (int job = wave; job < 49; job += 8) {
    int mi = job / 7, nj = job % 7;
    Frag a, bf;
    const uint4* rq = (const uint4*)&qs[lm + 16 * mi][0];
    a.q[0] = rq[half];
    a.q[1] = rq[2 + half];
    const uint4* rk = (const uint4*)&kk[lm + 16 * nj][0];
    bf.q[0] = rk[half];
    bf.q[1] = rk[2 + half];
    v8f acc = {0.f, 0.f, 0.f, 0.f, 0.f, 0.f, 0.f, 0.f};
    acc = wmma_bf16(a.v, bf.v, acc);
    for (int r = 0; r < 8; ++r) {
      int i = r + 8 * half + 16 * mi;
      sb[lm + 16 * nj][i] = f2bf(acc[r]);
    }
  }
  __syncthreads();

  // --- softmax over i (per column j), mask padded rows i>=102 ---
  for (int j = tid; j < CPAD; j += 256) {
    unsigned short* row = &sb[j][0];
    float mx = -1e30f;
    for (int i = 0; i < CC; ++i) mx = fmaxf(mx, bf2f(row[i]));
    float sum = 0.f;
    for (int i = 0; i < CC; ++i) {
      float e = __expf(bf2f(row[i]) - mx);
      sum += e;
      row[i] = f2bf(e);
    }
    float inv = 1.f / sum;
    for (int i = 0; i < CC; ++i) row[i] = f2bf(bf2f(row[i]) * inv);
    for (int i = CC; i < 128; ++i) row[i] = 0;
  }
  __syncthreads();

  // --- o[f][j] = gamma * sum_i v[f][i] beta[i][j] + xt[f][j];  write refined ---
  const float gamma = gamma_p[0];
  for (int job = wave; job < 14; job += 8) {
    int mt = job / 7, nt = job % 7;
    const uint4* ra = (const uint4*)&vs[lm + 16 * mt][0];
    const uint4* rb = (const uint4*)&sb[lm + 16 * nt][0];
    v8f acc = {0.f, 0.f, 0.f, 0.f, 0.f, 0.f, 0.f, 0.f};
    for (int ks = 0; ks < 4; ++ks) {
      Frag a, bf;
      a.q[0] = ra[half + 4 * ks];
      a.q[1] = ra[2 + half + 4 * ks];
      bf.q[0] = rb[half + 4 * ks];
      bf.q[1] = rb[2 + half + 4 * ks];
      acc = wmma_bf16(a.v, bf.v, acc);
    }
    int c = lm + 16 * nt;
    if (c < CC) {
      for (int r = 0; r < 8; ++r) {
        int f = r + 8 * half + 16 * mt;
        float val = gamma * acc[r] + bf2f(xts[c][f]);
        R[((long long)(b * CC + c)) * KFC + f * TT + t] = f2bf(val);
      }
    }
  }
}

// =====================================================================
// 5) FC: y[b][c][o] = R[b][c][:] . wfc[o][:] + bfc[o]  -> split output
// =====================================================================
__global__ void __launch_bounds__(128) k_fc(const unsigned short* __restrict__ R,
                                            const unsigned short* __restrict__ ffc,
                                            const float* __restrict__ bfc,
                                            float* __restrict__ out) {
  const int wave = threadIdx.x >> 5, lane = threadIdx.x & 31;
  const int half = lane >> 4, lm = lane & 15;
  const long long tile = (long long)blockIdx.x * 4 + wave;
  const long long row0 = tile * 16;
  const uint4* arow = (const uint4*)(R + (row0 + lm) * KFC);
  v8f z = {0.f, 0.f, 0.f, 0.f, 0.f, 0.f, 0.f, 0.f};
  v8f acc[4] = {z, z, z, z};
  for (int ks = 0; ks < 7; ++ks) {
    Frag a;
    a.q[0] = arow[half + 4 * ks];
    a.q[1] = arow[2 + half + 4 * ks];
    for (int nt = 0; nt < 4; ++nt) {
      Frag bf;
      const uint4* p = (const uint4*)(ffc + ((ks * 4 + nt) * 32 + lane) * 16);
      bf.q[0] = p[0];
      bf.q[1] = p[1];
      acc[nt] = wmma_bf16(a.v, bf.v, acc[nt]);
    }
  }
  for (int nt = 0; nt < 4; ++nt) {
    int o = lm + 16 * nt;
    float bb = bfc[o];
    for (int r = 0; r < 8; ++r) {
      long long row = row0 + r + 8 * half;      // b*C + c
      int b = (int)(row / CC), c = (int)(row % CC);
      int part = (c < 51) ? 0 : 1;
      long long idx = (long long)part * BB * 51 * 64 +
                      ((long long)b * 51 + (c - part * 51)) * 64 + o;
      out[idx] = acc[nt][r] + bb;
    }
  }
}

// =====================================================================
// launch
// =====================================================================
extern "C" void kernel_launch(void* const* d_in, const int* in_sizes, int n_in,
                              void* d_out, int out_size, void* d_ws,
                              size_t ws_size, hipStream_t stream) {
  (void)in_sizes; (void)n_in; (void)out_size; (void)ws_size;
  const float* x   = (const float*)d_in[0];
  const float* w1  = (const float*)d_in[1];
  const float* b1  = (const float*)d_in[2];
  const float* w2  = (const float*)d_in[3];
  const float* b2  = (const float*)d_in[4];
  const float* w3  = (const float*)d_in[5];
  const float* b3  = (const float*)d_in[6];
  const float* w4  = (const float*)d_in[7];
  const float* b4  = (const float*)d_in[8];
  const float* wq  = (const float*)d_in[9];
  const float* bq  = (const float*)d_in[10];
  const float* wk  = (const float*)d_in[11];
  const float* bk  = (const float*)d_in[12];
  const float* wv  = (const float*)d_in[13];
  const float* bv  = (const float*)d_in[14];
  const float* gm  = (const float*)d_in[15];
  const float* wfc = (const float*)d_in[16];
  const float* bfc = (const float*)d_in[17];
  float* out = (float*)d_out;

  char* ws = (char*)d_ws;
  unsigned short* frags = (unsigned short*)(ws + OFF_FRAGS);
  float*          xT    = (float*)(ws + OFF_XT);
  unsigned short* h2    = (unsigned short*)(ws + OFF_H2);
  unsigned short* h3    = (unsigned short*)(ws + OFF_H3);
  unsigned short* h4    = (unsigned short*)(ws + OFF_H4);
  unsigned short* R     = (unsigned short*)(ws + OFF_R);

  const int pairs = BB * CC;                    // 52224
  k_transpose<<<(BB * LL * CC + 255) / 256, 256, 0, stream>>>(x, xT);
  k_repack<<<1, 256, 0, stream>>>(w2, w3, w4, wq, wk, wv, wfc, frags);
  k_conv12<<<pairs / 4, 128, 0, stream>>>(xT, w1, b1, b2, frags, h2);
  k_conv<39, 18, 2, 1><<<pairs / 4, 128, 0, stream>>>(h2, h3, frags, b3);
  k_conv<18, 7, 1, 2><<<pairs / 4, 128, 0, stream>>>(h3, h4, frags, b4);
  k_attn<<<BB * TT, 256, 0, stream>>>(h4, frags + 15360, bq, bk, bv, gm, R);
  k_fc<<<(pairs / 16) / 4, 128, 0, stream>>>(R, frags + 18432, bfc, out);
}
